// LTAE1d_64115271794771
// MI455X (gfx1250) — matrix operand order repaired and checked by reading, not compiled, not run on hardware
//
#include <hip/hip_runtime.h>
#include <hip/hip_bf16.h>
#include <math.h>

// ---- shapes (fixed by reference) ----
// B=1024, T=60 (padded 64), IN_CH=10, DM=256, H=16, DK=4, A=13, DH=16, MLP_OUT=128
#define NB   1024
#define TT   60
#define TP   64
#define DM   256
#define NH   16
#define NA   13
#define MOUT 128
#define AB   (NA*NB)        // 13312
#define HBA  (NB*NA)        // 13312 (B*A)
#define ROWS 208            // A*H rows per b2 block = 13 M-tiles of 16

typedef __attribute__((ext_vector_type(16))) _Float16 v16h;
typedef __attribute__((ext_vector_type(8)))  _Float16 v8h;
typedef __attribute__((ext_vector_type(8)))  float    v8f;
typedef __attribute__((ext_vector_type(4)))  float    f4;
typedef __attribute__((ext_vector_type(4)))  int      v4i;

#if __has_builtin(__builtin_amdgcn_global_load_async_to_lds_b128)
#define HAVE_ASYNC_LOAD 1
#else
#define HAVE_ASYNC_LOAD 0
#endif
#if __has_builtin(__builtin_amdgcn_global_store_async_from_lds_b128)
#define HAVE_ASYNC_STORE 1
#else
#define HAVE_ASYNC_STORE 0
#endif

// async-to-LDS builtin param types: (int4 AS1*, int4 AS3*, imm, imm)
typedef __attribute__((address_space(1))) v4i gas_v4i;
typedef __attribute__((address_space(3))) v4i las_v4i;

static __device__ __forceinline__ void wait_async0() {
#if __has_builtin(__builtin_amdgcn_s_wait_asynccnt)
  __builtin_amdgcn_s_wait_asynccnt(0);
#else
  asm volatile("s_wait_asynccnt 0" ::: "memory");
#endif
}

// A-fragment (16x32 f16): lane halves take contiguous K {0..15} / {16..31}
static __device__ __forceinline__ v16h load_a16(const _Float16* rowbase, int koff) {
  v8h lo = *(const v8h*)(rowbase + koff);
  v8h hi = *(const v8h*)(rowbase + koff + 8);
  v16h a;
#pragma unroll
  for (int j = 0; j < 8; ++j) { a[j] = lo[j]; a[j + 8] = hi[j]; }
  return a;
}

// =====================================================================
// K1: inconv (1x1 conv) + GroupNorm(H=16) fused. One block per sample b.
// Thread d owns channel d; h in registers; stats via wave32 shfl_xor;
// normalized tile staged in LDS, then written out with async LDS->global
// b128 stores (fallback: 16B vector stores).
// =====================================================================
__global__ __launch_bounds__(256) void k1_inconv_gn(
    const float* __restrict__ x, const float* __restrict__ Wc,
    const float* __restrict__ bc, const float* __restrict__ gnw,
    const float* __restrict__ gnb, _Float16* __restrict__ vh)
{
  __shared__ float sx[TT * 10];
  __shared__ _Float16 sv[TP * DM];
  __shared__ float smu[NH], srs[NH];
  const int b = blockIdx.x, d = threadIdx.x;

  for (int idx = d; idx < TT * 10; idx += 256) sx[idx] = x[b * (TT * 10) + idx];
  __syncthreads();

  float wrow[10];
#pragma unroll
  for (int c = 0; c < 10; ++c) wrow[c] = Wc[d * 10 + c];
  const float bias = bc[d];

  float hv[TT];
  float s = 0.f, s2 = 0.f;
#pragma unroll
  for (int t = 0; t < TT; ++t) {
    float acc = bias;
#pragma unroll
    for (int c = 0; c < 10; ++c) acc = fmaf(sx[t * 10 + c], wrow[c], acc);
    hv[t] = acc; s += acc; s2 = fmaf(acc, acc, s2);
  }
  // group g = d>>4 spans 16 consecutive lanes of a wave-half -> xor 1,2,4,8
#pragma unroll
  for (int off = 1; off < 16; off <<= 1) {
    s  += __shfl_xor(s,  off, 32);
    s2 += __shfl_xor(s2, off, 32);
  }
  const int g = d >> 4;
  if ((d & 15) == 0) {
    float mu = s * (1.f / 960.f);
    float var = s2 * (1.f / 960.f) - mu * mu;
    smu[g] = mu; srs[g] = rsqrtf(var + 1e-5f);
  }
  __syncthreads();
  const float mu = smu[g], rs = srs[g];
  const float gw = gnw[d], gb = gnb[d];
#pragma unroll
  for (int t = 0; t < TT; ++t)
    sv[t * DM + d] = (_Float16)(((hv[t] - mu) * rs) * gw + gb);
#pragma unroll
  for (int t = TT; t < TP; ++t) sv[t * DM + d] = (_Float16)0.f;
  __syncthreads();

  _Float16* vb = vh + (size_t)b * (TP * DM);
#if HAVE_ASYNC_STORE
  for (int idx = d * 8; idx < TP * DM; idx += 256 * 8)
    __builtin_amdgcn_global_store_async_from_lds_b128(
        (gas_v4i*)(vb + idx), (las_v4i*)(&sv[idx]), 0, 0);
  // S_ENDPGM performs an implicit wait-idle; no explicit drain needed.
#else
  for (int idx = d * 8; idx < TP * DM; idx += 256 * 8)
    *(f4*)(&vb[idx]) = *(const f4*)(&sv[idx]);
#endif
}

// =====================================================================
// K2: W1 -> f16
// =====================================================================
__global__ __launch_bounds__(256) void k2_cvt_w1(const float* __restrict__ W1,
                                                 _Float16* __restrict__ w1h)
{
  int i = blockIdx.x * 256 + threadIdx.x;
  if (i < MOUT * DM) w1h[i] = (_Float16)W1[i];
}

// =====================================================================
// K3: fused attention per b2. 128 threads = 4 waves (wave = N-tile over t).
//   Weff[n] = Q[a1,h1,:] . Wk[(a2*16+h2)*4 : +4, :]  (DK=4 folded into W),
//   logits[208][64] = Weff x V[b2]^T via v_wmma_f32_16x16x32_f16,
//   softmax over t<60 (x0.5 = 1/sqrt(DK); bk cancels in softmax exactly),
//   out = attn @ v-head, vectorized row-per-thread, scattered by (a1,h1,b1).
// =====================================================================
__global__ __launch_bounds__(128) void k3_attn(
    const _Float16* __restrict__ vh, const float* __restrict__ Q,
    const float* __restrict__ Wk, _Float16* __restrict__ out2)
{
  __shared__ _Float16 svt[TP * DM];       // 32 KB v tile (t-padded)
  __shared__ _Float16 schA[ROWS * 32];    // 13 KB Weff K-chunk
  __shared__ float    sL[ROWS * TP];      // 52 KB logits/probs

  const int b2 = blockIdx.x, tid = threadIdx.x;
  const int lane = tid & 31, w = tid >> 5;

  // v tile: async global->LDS (overlaps with first Weff chunk build)
  const _Float16* vsrc = vh + (size_t)b2 * (TP * DM);
#if HAVE_ASYNC_LOAD
  for (int idx = tid * 8; idx < TP * DM; idx += 128 * 8)
    __builtin_amdgcn_global_load_async_to_lds_b128(
        (gas_v4i*)(vsrc + idx), (las_v4i*)(&svt[idx]), 0, 0);
#else
  for (int idx = tid * 8; idx < TP * DM; idx += 128 * 8)
    *(f4*)(&svt[idx]) = *(const f4*)(&vsrc[idx]);
#endif

  v8f acc[13] = {};
  const int col0 = lane & 15;
  const int koff = (lane < 16) ? 0 : 16;  // A-frag K sub-offset
  const int kb   = (lane < 16) ? 0 : 16;  // B-frag K sub-offset

  for (int ks = 0; ks < 8; ++ks) {        // K = 256 in 8 steps of 32
    __syncthreads();                      // schA reuse guard
    // Build Weff chunk: row i = h2*13 + a2
#pragma unroll
    for (int r = 0; r < 2; ++r) {
      int i = tid + r * 128;
      if (i < ROWS) {
        int h2 = i / 13, a2 = i - h2 * 13;
        int n  = h2 * HBA + b2 * NA + a2;
        int a1 = n >> 14, h1 = (n >> 10) & 15;
        const float* qp = Q + (a1 * 64 + h1 * 4);
        float q0 = qp[0], q1 = qp[1], q2 = qp[2], q3 = qp[3];
        const float* wk = Wk + ((a2 * NH + h2) * 4) * DM + ks * 32;
        if (ks < 7) {                     // prefetch next K-chunk of Wk (L2 stream)
          __builtin_prefetch(wk + 32, 0, 3);
          __builtin_prefetch(wk + DM + 32, 0, 3);
          __builtin_prefetch(wk + 2 * DM + 32, 0, 3);
          __builtin_prefetch(wk + 3 * DM + 32, 0, 3);
        }
#pragma unroll
        for (int kk = 0; kk < 32; ++kk) {
          float val = fmaf(q0, wk[kk],
                      fmaf(q1, wk[DM + kk],
                      fmaf(q2, wk[2 * DM + kk], q3 * wk[3 * DM + kk])));
          schA[i * 32 + kk] = (_Float16)val;
        }
      }
    }
#if HAVE_ASYNC_LOAD
    if (ks == 0) wait_async0();           // v tile landed while chunk 0 was built
#endif
    __syncthreads();

    // B fragment: col = time t, K from v features
    const int tcol = (w << 4) + col0;
    v16h bfrag = *(const v16h*)(&svt[tcol * DM + ks * 32 + kb]);

    // software-pipelined A-fragment loads
    v16h acur = load_a16(&schA[col0 * 32], koff);
#pragma unroll
    for (int mt = 0; mt < 13; ++mt) {
      v16h anext = acur;
      if (mt < 12) anext = load_a16(&schA[(((mt + 1) << 4) + col0) * 32], koff);
      acc[mt] = __builtin_amdgcn_wmma_f32_16x16x32_f16(
          false, acur, false, bfrag, (short)0, acc[mt], false, false);
      acur = anext;
    }
  }

  // D layout: VGPR r -> rows r / r+8; col = lane&15 within the N-tile
  {
    const int cb = (w << 4) + col0;
    const int rofs = (lane < 16) ? 0 : 8;
#pragma unroll
    for (int mt = 0; mt < 13; ++mt)
#pragma unroll
      for (int r = 0; r < 8; ++r)
        sL[((mt << 4) + rofs + r) * TP + cb] = acc[mt][r];
  }
  __syncthreads();

  // softmax over t<60 (scale by 1/TEMP = 0.5)
#pragma unroll
  for (int r = 0; r < 2; ++r) {
    int i = tid + r * 128;
    if (i < ROWS) {
      float* row = sL + i * TP;
      float mx = -3.4e38f;
      for (int t = 0; t < TT; ++t) { float lv = row[t] * 0.5f; row[t] = lv; mx = fmaxf(mx, lv); }
      float sum = 0.f;
      for (int t = 0; t < TT; ++t) { float e = __expf(row[t] - mx); row[t] = e; sum += e; }
      float inv = 1.f / sum;
      for (int t = 0; t < TT; ++t) row[t] *= inv;
    }
  }
  __syncthreads();

  // attn @ v-head: row-per-thread, vectorized v-row loads + 32B store
#pragma unroll
  for (int r = 0; r < 2; ++r) {
    int i = tid + r * 128;
    if (i < ROWS) {
      int h2 = i / 13, a2 = i - h2 * 13;
      int n  = h2 * HBA + b2 * NA + a2;
      int a1 = n >> 14, h1 = (n >> 10) & 15, bi1 = n & 1023;
      const float* prow = sL + i * TP;
      float oacc[16] = {};
      for (int t = 0; t < TT; ++t) {
        float p = prow[t];
        v8h va = *(const v8h*)(&svt[t * DM + (h2 << 4)]);
        v8h vb = *(const v8h*)(&svt[t * DM + (h2 << 4) + 8]);
#pragma unroll
        for (int j = 0; j < 8; ++j) {
          oacc[j]     = fmaf(p, (float)va[j], oacc[j]);
          oacc[j + 8] = fmaf(p, (float)vb[j], oacc[j + 8]);
        }
      }
      v16h ov;
#pragma unroll
      for (int j = 0; j < 16; ++j) ov[j] = (_Float16)oacc[j];
      *(v16h*)(&out2[((size_t)(a1 << 10) + bi1) * DM + (h1 << 4)]) = ov;
    }
  }
}

// =====================================================================
// K4: MLP GEMM (WMMA) + BN(eval) + ReLU + GroupNorm(groups of 8) + transpose
// 128 threads = 4 waves; block covers 64 rows of out2 (M=13312 = 208 blocks)
// =====================================================================
__global__ __launch_bounds__(128) void k4_mlp(
    const _Float16* __restrict__ out2, const _Float16* __restrict__ w1h,
    const float* __restrict__ lb1, const float* __restrict__ bnw,
    const float* __restrict__ bnb, const float* __restrict__ bnrm,
    const float* __restrict__ bnrv, const float* __restrict__ gow,
    const float* __restrict__ gob, float* __restrict__ out)
{
  const int tid = threadIdx.x, lane = tid & 31, w = tid >> 5;
  const int m0 = blockIdx.x * 64 + w * 16;
  const int col0 = lane & 15;
  const int koff = (lane < 16) ? 0 : 16;
  const int kb   = (lane < 16) ? 0 : 16;

  v8f acc[8] = {};
  const _Float16* arow = out2 + (size_t)(m0 + col0) * DM;
  v16h afrag = load_a16(arow, koff);
  for (int ks = 0; ks < 8; ++ks) {
    v16h acur = afrag;
    if (ks < 7) afrag = load_a16(arow + (ks + 1) * 32, koff);
#pragma unroll
    for (int nt = 0; nt < 8; ++nt) {
      v16h bfrag = *(const v16h*)(&w1h[(nt * 16 + col0) * DM + ks * 32 + kb]);
      acc[nt] = __builtin_amdgcn_wmma_f32_16x16x32_f16(
          false, acur, false, bfrag, (short)0, acc[nt], false, false);
    }
  }

  const int rofs = (lane < 16) ? 0 : 8;
#pragma unroll
  for (int nt = 0; nt < 8; ++nt) {
    const int col = nt * 16 + col0;
    const float sc = bnw[col] * rsqrtf(bnrv[col] + 1e-5f);
    const float sh = bnb[col] - bnrm[col] * sc;
    const float bi = lb1[col];
    const float gw = gow[col], gb = gob[col];
#pragma unroll
    for (int r = 0; r < 8; ++r)
      acc[nt][r] = fmaxf(fmaf(acc[nt][r] + bi, sc, sh), 0.f);
    // GroupNorm over 8 cols (same row): xor 1,2,4 stays inside the 8-lane group
#pragma unroll
    for (int r = 0; r < 8; ++r) {
      float y = acc[nt][r];
      float sm = y, s2 = y * y;
#pragma unroll
      for (int off = 1; off < 8; off <<= 1) {
        sm += __shfl_xor(sm, off, 32);
        s2 += __shfl_xor(s2, off, 32);
      }
      float mu  = sm * 0.125f;
      float var = s2 * 0.125f - mu * mu;
      float yn  = (y - mu) * rsqrtf(var + 1e-5f) * gw + gb;
      int m = m0 + rofs + r;
      int a = m >> 10, bb = m & 1023;
      out[((size_t)(bb * NA + a)) * MOUT + col] = yn;
    }
  }
}

// =====================================================================
extern "C" void kernel_launch(void* const* d_in, const int* in_sizes, int n_in,
                              void* d_out, int out_size, void* d_ws, size_t ws_size,
                              hipStream_t stream)
{
  const float* x    = (const float*)d_in[0];
  const float* Wc   = (const float*)d_in[1];
  const float* bc   = (const float*)d_in[2];
  const float* gniw = (const float*)d_in[3];
  const float* gnib = (const float*)d_in[4];
  const float* Q    = (const float*)d_in[5];
  const float* Wk   = (const float*)d_in[6];
  // d_in[7] = bk: constant over t per attention row -> cancels in softmax
  const float* W1   = (const float*)d_in[8];
  const float* b1v  = (const float*)d_in[9];
  const float* bnw  = (const float*)d_in[10];
  const float* bnb  = (const float*)d_in[11];
  const float* bnrm = (const float*)d_in[12];
  const float* bnrv = (const float*)d_in[13];
  const float* gow  = (const float*)d_in[14];
  const float* gob  = (const float*)d_in[15];
  float* out = (float*)d_out;

  const size_t vh_bytes   = (size_t)NB * TP * DM * sizeof(_Float16);  // 33.5 MB
  const size_t out2_bytes = (size_t)AB * DM * sizeof(_Float16);       // 6.8 MB
  const size_t w1h_bytes  = (size_t)MOUT * DM * sizeof(_Float16);
  if (ws_size < vh_bytes + out2_bytes + w1h_bytes) return;

  _Float16* vh_buf = (_Float16*)d_ws;
  _Float16* out2   = (_Float16*)((char*)d_ws + vh_bytes);
  _Float16* w1h    = (_Float16*)((char*)d_ws + vh_bytes + out2_bytes);

  k1_inconv_gn<<<NB, 256, 0, stream>>>(x, Wc, bc, gniw, gnib, vh_buf);
  k2_cvt_w1<<<(MOUT * DM + 255) / 256, 256, 0, stream>>>(W1, w1h);
  k3_attn<<<NB, 128, 0, stream>>>(vh_buf, Q, Wk, out2);
  k4_mlp<<<AB / 64, 128, 0, stream>>>(out2, w1h, b1v, bnw, bnb, bnrm, bnrv,
                                      gow, gob, out);
  (void)in_sizes; (void)n_in; (void)out_size;
}
